// SWACrossConditionalSelfAttention_22832046145859
// MI455X (gfx1250) — matrix operand order repaired and checked by reading, not compiled, not run on hardware
//
#include <hip/hip_runtime.h>

typedef __attribute__((ext_vector_type(16))) __bf16 bfrag;
typedef __attribute__((ext_vector_type(8)))  __bf16 bhalf8;
typedef __attribute__((ext_vector_type(8)))  float  v8f;
typedef __attribute__((ext_vector_type(4)))  float  v4f;
typedef __attribute__((ext_vector_type(4)))  __bf16 bhalf4;

constexpr int TT  = 87;    // sequence length
constexpr int Tp  = 96;    // padded to 6 tiles of 16
constexpr int Cc  = 512;
constexpr int Hh  = 8;
constexpr int Dd  = 64;
constexpr int Bb  = 512;
constexpr int BT  = Bb * TT;        // 44544
constexpr int MT  = BT / 16;        // 2784 M-tiles (exact)

// ---------------- fragment loaders (ISA 7.12.2 layouts) ----------------

// A 16x32 bf16 fragment. lane=(m = l&15, h = l>>4)
// element e -> row m, col kbase + (e/8)*16 + h*8 + (e%8)   (two contiguous 8-elem chunks)
__device__ inline bfrag a_frag_bf16(const __bf16* __restrict__ row0, int ld, int lane, int kbase) {
  int m = lane & 15, h = lane >> 4;
  const __bf16* p = row0 + (size_t)m * ld + kbase + h * 8;
  bhalf8 lo = *(const bhalf8*)p;         // 16B -> b128
  bhalf8 hi = *(const bhalf8*)(p + 16);  // 16B -> b128
  bfrag a;
#pragma unroll
  for (int e = 0; e < 8; ++e) { a[e] = lo[e]; a[e + 8] = hi[e]; }
  return a;
}

// B 32x16 bf16 fragment where B[k][n] = stored[n][k] (stored row-major, ld elems).
// lane=(n = l&15, h = l>>4): element e -> K = kbase + (e/8)*16 + h*8 + (e%8), col n. two 8-elem chunks.
__device__ inline bfrag b_frag_bf16(const __bf16* __restrict__ stored, int ld, int lane, int kbase) {
  int n = lane & 15, h = lane >> 4;
  const __bf16* p = stored + (size_t)n * ld + kbase + h * 8;
  bhalf8 lo = *(const bhalf8*)p;
  bhalf8 hi = *(const bhalf8*)(p + 16);
  bfrag b;
#pragma unroll
  for (int e = 0; e < 8; ++e) { b[e] = lo[e]; b[e + 8] = hi[e]; }
  return b;
}

__device__ inline v8f wmma_bf16(bfrag a, bfrag b, v8f c) {
  return __builtin_amdgcn_wmma_f32_16x16x32_bf16(false, a, false, b, (short)0, c, false, false);
}

// ---------------- setup kernels ----------------

// vectorized fp32 -> bf16 conversion (4 elems/thread: one b128 load, one b64 store)
__global__ void cvt_kernel(const float* __restrict__ w, __bf16* __restrict__ o, int n4) {
  int i = blockIdx.x * blockDim.x + threadIdx.x;
  if (i < n4) {
    v4f f = *(const v4f*)(w + (size_t)i * 4);
    bhalf4 b;
#pragma unroll
    for (int e = 0; e < 4; ++e) b[e] = (__bf16)f[e];
    *(bhalf4*)(o + (size_t)i * 4) = b;
  }
}

__global__ void zero_kernel(unsigned int* __restrict__ p, size_t n32) {
  size_t i = (size_t)blockIdx.x * blockDim.x + threadIdx.x;
  if (i < n32) p[i] = 0u;
}

// ---------------- fused QKV projection ----------------
// grid (MT, 8 heads, 3 mats), block 32. Wave computes 16 rows x 64 cols, K=512.
// All operands pre-converted bf16: inner loop is pure b128 loads + WMMA.
__global__ void qkv_kernel(const __bf16* __restrict__ xb,
                           const float* __restrict__ bq, const float* __restrict__ bk,
                           const float* __restrict__ bv,
                           const __bf16* __restrict__ Wqb, const __bf16* __restrict__ Wkb,
                           const __bf16* __restrict__ Wvb,
                           __bf16* __restrict__ qd, __bf16* __restrict__ kd,
                           __bf16* __restrict__ vTd) {
  const int lane = threadIdx.x;
  const int mt = blockIdx.x;
  const int nh = blockIdx.y;     // 64-wide col block == head
  const int mat = blockIdx.z;    // 0=q 1=k 2=v
  const __bf16* Wb  = (mat == 0) ? Wqb : (mat == 1) ? Wkb : Wvb;
  const float*  bia = (mat == 0) ? bq  : (mat == 1) ? bk  : bv;

  v8f acc[4] = {};
  const __bf16* xrow = xb + (size_t)mt * 16 * Cc;
#pragma unroll 4
  for (int ks = 0; ks < Cc / 32; ++ks) {
    bfrag a = a_frag_bf16(xrow, Cc, lane, ks * 32);
#pragma unroll
    for (int jc = 0; jc < 4; ++jc) {
      bfrag b = b_frag_bf16(Wb + (size_t)(nh * 64 + jc * 16) * Cc, Cc, lane, ks * 32);
      acc[jc] = wmma_bf16(a, b, acc[jc]);
    }
  }

  const int n = lane & 15, h = lane >> 4;
#pragma unroll
  for (int jc = 0; jc < 4; ++jc) {
    int d = jc * 16 + n;
    float bval = bia[nh * 64 + d];
#pragma unroll
    for (int r = 0; r < 8; ++r) {
      int g = mt * 16 + r + 8 * h;   // global row in [B*T]
      int bbi = g / TT;
      int t = g - bbi * TT;
      float val = acc[jc][r] + bval;
      if (mat < 2) {
        __bf16* dst = (mat == 0) ? qd : kd;
        dst[(((size_t)bbi * Hh + nh) * Tp + t) * Dd + d] = (__bf16)val;   // [B,H,Tp,D]
      } else {
        vTd[(((size_t)bbi * Hh + nh) * Dd + d) * Tp + t] = (__bf16)val;   // [B,H,D,Tp]
      }
    }
  }
}

// ---------------- attention: S = qk^T*scale, banded mask, softmax, Y = P v ----------------
// grid (B*H), block 32 (one wave per (b,h)).
__global__ void attn_kernel(const __bf16* __restrict__ q, const __bf16* __restrict__ k,
                            const __bf16* __restrict__ vT, __bf16* __restrict__ y) {
  __shared__ float Sl[16 * Tp];                     // one row-tile of scores, f32
  __shared__ __align__(16) __bf16 Pl[Tp * Tp];     // softmaxed probabilities, bf16

  const int lane = threadIdx.x;
  const int n = lane & 15, h = lane >> 4;
  const int bh = blockIdx.x;
  const int bbi = bh >> 3, head = bh & 7;
  const __bf16* qp = q  + (size_t)bh * Tp * Dd;
  const __bf16* kp = k  + (size_t)bh * Tp * Dd;
  const __bf16* vp = vT + (size_t)bh * Dd * Tp;
  const float scale = 0.125f;                       // 1/sqrt(64)

  for (int i = 0; i < 6; ++i) {
    bfrag a0 = a_frag_bf16(qp + i * 16 * Dd, Dd, lane, 0);
    bfrag a1 = a_frag_bf16(qp + i * 16 * Dd, Dd, lane, 32);
#pragma unroll 1
    for (int j = 0; j < 6; ++j) {
      bfrag b0 = b_frag_bf16(kp + j * 16 * Dd, Dd, lane, 0);
      bfrag b1 = b_frag_bf16(kp + j * 16 * Dd, Dd, lane, 32);
      v8f c = {};
      c = wmma_bf16(a0, b0, c);
      c = wmma_bf16(a1, b1, c);
      int col = j * 16 + n;
      int jj = col % 29;
#pragma unroll
      for (int r = 0; r < 8; ++r) {
        int row = i * 16 + r + 8 * h;
        int ii = row % 29;
        // allowed iff col<87 and (j'<=i') and (i'-j' < WINDOW=22)  (3x3 tiling of 29x29 band)
        bool allowed = (col < TT) && (jj <= ii) && (ii - jj < 22);
        float s = c[r] * scale;
        Sl[(r + 8 * h) * Tp + col] = allowed ? s : -1e30f;
      }
    }
    __syncthreads();
    // softmax: 16 rows, two lanes per row (lane pair {n, n+16}), 48 cols each
    {
      const float* srow = &Sl[n * Tp + h * 48];
      float mx = -3e30f;
      for (int cc = 0; cc < 48; ++cc) mx = fmaxf(mx, srow[cc]);
      mx = fmaxf(mx, __shfl_xor(mx, 16, 32));
      float sum = 0.f;
      for (int cc = 0; cc < 48; ++cc) sum += __expf(srow[cc] - mx);
      sum += __shfl_xor(sum, 16, 32);
      float inv = 1.0f / sum;
      __bf16* prow = &Pl[(i * 16 + n) * Tp + h * 48];
      for (int cc = 0; cc < 48; ++cc) prow[cc] = (__bf16)(__expf(srow[cc] - mx) * inv);
    }
    __syncthreads();
  }

  // Y = P @ V : A = P (from LDS), B = v via vT (contiguous)
  for (int i = 0; i < 6; ++i) {
    bfrag pa[3];
#pragma unroll
    for (int ks = 0; ks < 3; ++ks) pa[ks] = a_frag_bf16(&Pl[i * 16 * Tp], Tp, lane, ks * 32);
#pragma unroll 1
    for (int jd = 0; jd < 4; ++jd) {
      v8f c = {};
#pragma unroll
      for (int ks = 0; ks < 3; ++ks) {
        bfrag b = b_frag_bf16(vp + (size_t)jd * 16 * Tp, Tp, lane, ks * 32);
        c = wmma_bf16(pa[ks], b, c);
      }
      int d = jd * 16 + n;
#pragma unroll
      for (int r = 0; r < 8; ++r) {
        int t = i * 16 + r + 8 * h;
        if (t < TT)
          y[(((size_t)bbi * TT + t) * Hh + head) * Dd + d] = (__bf16)c[r];  // [B,T,C]
      }
    }
  }
}

// ---------------- output projection: out = y @ Wp^T + bp (fp32 out) ----------------
__global__ void proj_kernel(const __bf16* __restrict__ yb, const __bf16* __restrict__ Wpb,
                            const float* __restrict__ bp, float* __restrict__ out) {
  const int lane = threadIdx.x;
  const int mt = blockIdx.x, ny = blockIdx.y;
  v8f acc[4] = {};
  const __bf16* yrow = yb + (size_t)mt * 16 * Cc;
#pragma unroll 4
  for (int ks = 0; ks < Cc / 32; ++ks) {
    bfrag a = a_frag_bf16(yrow, Cc, lane, ks * 32);
#pragma unroll
    for (int jc = 0; jc < 4; ++jc) {
      bfrag b = b_frag_bf16(Wpb + (size_t)(ny * 64 + jc * 16) * Cc, Cc, lane, ks * 32);
      acc[jc] = wmma_bf16(a, b, acc[jc]);
    }
  }
  const int n = lane & 15, h = lane >> 4;
#pragma unroll
  for (int jc = 0; jc < 4; ++jc) {
    int col = ny * 64 + jc * 16 + n;
    float bval = bp[col];
#pragma unroll
    for (int r = 0; r < 8; ++r) {
      int g = mt * 16 + r + 8 * h;
      out[(size_t)g * Cc + col] = acc[jc][r] + bval;
    }
  }
}

// ---------------- host ----------------
extern "C" void kernel_launch(void* const* d_in, const int* in_sizes, int n_in,
                              void* d_out, int out_size, void* d_ws, size_t ws_size,
                              hipStream_t stream) {
  const float* x  = (const float*)d_in[0];
  const float* Wq = (const float*)d_in[1];
  const float* bq = (const float*)d_in[2];
  const float* Wk = (const float*)d_in[3];
  const float* bk = (const float*)d_in[4];
  const float* Wv = (const float*)d_in[5];
  const float* bv = (const float*)d_in[6];
  const float* Wp = (const float*)d_in[7];
  const float* bp = (const float*)d_in[8];
  float* out = (float*)d_out;

  char* ws = (char*)d_ws;
  const size_t szW   = (size_t)Cc * Cc * sizeof(__bf16);           // 512 KB each
  const size_t szQKV = (size_t)Bb * Hh * Tp * Dd * sizeof(__bf16); // 48 MB each
  const size_t szX   = (size_t)BT * Cc * sizeof(__bf16);           // 45.6 MB
  size_t off = 0;
  __bf16* Wqb = (__bf16*)(ws + off); off += szW;
  __bf16* Wkb = (__bf16*)(ws + off); off += szW;
  __bf16* Wvb = (__bf16*)(ws + off); off += szW;
  __bf16* Wpb = (__bf16*)(ws + off); off += szW;
  __bf16* xb  = (__bf16*)(ws + off); off += szX;
  __bf16* qd  = (__bf16*)(ws + off); off += szQKV;
  __bf16* kd  = (__bf16*)(ws + off); off += szQKV;
  __bf16* vTd = (__bf16*)(ws + off); off += szQKV;
  __bf16* yb  = (__bf16*)(ws + off); off += szX;

  const int nw4 = Cc * Cc / 4;
  cvt_kernel<<<(nw4 + 255) / 256, 256, 0, stream>>>(Wq, Wqb, nw4);
  cvt_kernel<<<(nw4 + 255) / 256, 256, 0, stream>>>(Wk, Wkb, nw4);
  cvt_kernel<<<(nw4 + 255) / 256, 256, 0, stream>>>(Wv, Wvb, nw4);
  cvt_kernel<<<(nw4 + 255) / 256, 256, 0, stream>>>(Wp, Wpb, nw4);
  const int nx4 = BT * Cc / 4;
  cvt_kernel<<<(nx4 + 255) / 256, 256, 0, stream>>>(x, xb, nx4);

  size_t n32 = 3 * szQKV / 4;  // zero q,k,vT (covers the Tp padding rows/cols)
  zero_kernel<<<(unsigned)((n32 + 255) / 256), 256, 0, stream>>>((unsigned int*)qd, n32);

  dim3 gq(MT, Hh, 3);
  qkv_kernel<<<gq, 32, 0, stream>>>(xb, bq, bk, bv, Wqb, Wkb, Wvb, qd, kd, vTd);

  attn_kernel<<<Bb * Hh, 32, 0, stream>>>(qd, kd, vTd, yb);

  dim3 gp(MT, Hh);
  proj_kernel<<<gp, 32, 0, stream>>>(yb, Wpb, bp, out);
}
